// FGST_66949950210148
// MI455X (gfx1250) — matrix operand order repaired and checked by reading, not compiled, not run on hardware
//
#include <hip/hip_runtime.h>
#include <hip/hip_bf16.h>
#include <stdint.h>

// ---------------------------------------------------------------------------
// FGSW-MSA on gfx1250, fused: warp+LN -> [conv_q + conv_kv + attention in one
// workgroup-per-window kernel, Q/K/V held in LDS (112KB of the 320KB WGP LDS)]
// -> output conv. All GEMMs via v_wmma_f32_16x16x32_f16; activations NHWC f16
// with tap-major K so every fragment is contiguous b128 loads.
// ---------------------------------------------------------------------------

typedef __attribute__((ext_vector_type(16))) _Float16 v16h;
typedef __attribute__((ext_vector_type(8)))  _Float16 v8h;
typedef __attribute__((ext_vector_type(8)))  float    v8f;

#define HH 128
#define WW 128
#define HW (HH*WW)
#define CIN 64
#define INNER 512
#define DHEAD 64
#define NWIN 1024          // 32x32 windows per image
#define QL 16
#define KVL 48

// ---- WMMA fragment index maps (contiguous k order used by this code, wave32) ------------------------
// A (16x32 f16): lane -> M = lane&15 ; one contiguous 16-elem run at K offset
//   boff=(lane>>4)*16
// B (32x16 f16): lane -> N = lane&15 ; one contiguous 16-elem run at (lane>>4)*16
// C/D (16x16 f32): lane -> N = lane&15 ; element r -> M = r + (lane>>4)*8

__device__ __forceinline__ v8f wmma_f16(v16h a, v16h b, v8f c) {
  return __builtin_amdgcn_wmma_f32_16x16x32_f16(false, a, false, b,
                                                (short)0, c, false, false);
}

template <typename P>
__device__ __forceinline__ v16h load_a_runs(P p) {   // two aligned 16B runs
  v8h r0 = *(const v8h*)(p);
  v8h r1 = *(const v8h*)(p + 8);
  return __builtin_shufflevector(r0, r1, 0,1,2,3,4,5,6,7,8,9,10,11,12,13,14,15);
}

// ---------------------------------------------------------------------------
// 0) weight repack fp32 (co,ci,3,3) -> f16 (co, tap, ci)   [k = tap*cin + ci]
// ---------------------------------------------------------------------------
__global__ void repack_w_kernel(const float* __restrict__ src,
                                _Float16* __restrict__ dst,
                                int cout, int cin) {
  int i = blockIdx.x * blockDim.x + threadIdx.x;
  int total = cout * cin * 9;
  if (i >= total) return;
  int co  = i / (cin * 9);
  int r   = i - co * cin * 9;     // = tap*cin + ci
  int tap = r / cin;
  int ci  = r - tap * cin;
  dst[i] = (_Float16)src[((size_t)(co * cin + ci)) * 9 + tap];
}

// ---------------------------------------------------------------------------
// 1) LayerNorm of q over channels -> f16 NHWC  qn[(n*HW+p)*64 + c]
// ---------------------------------------------------------------------------
__global__ void prep_q_kernel(const float* __restrict__ q_inp,
                              const float* __restrict__ g,
                              const float* __restrict__ b,
                              _Float16* __restrict__ qn) {
  int idx = blockIdx.x * blockDim.x + threadIdx.x;   // (n, p): 2*16384
  int n = idx >> 14;
  int p = idx & (HW - 1);
  const float* base = q_inp + (size_t)n * CIN * HW + p;
  float vals[CIN];
  float mu = 0.f;
#pragma unroll 8
  for (int c = 0; c < CIN; ++c) { vals[c] = base[(size_t)c * HW]; mu += vals[c]; }
  mu *= (1.0f / CIN);
  float var = 0.f;
#pragma unroll 8
  for (int c = 0; c < CIN; ++c) { float d = vals[c] - mu; var += d * d; }
  var *= (1.0f / CIN);
  float rs = rsqrtf(var + 1e-5f);
  _Float16* ob = qn + ((size_t)n * HW + p) * CIN;
#pragma unroll 8
  for (int c = 0; c < CIN; ++c)
    ob[c] = (_Float16)((vals[c] - mu) * rs * g[c] + b[c]);
}

// ---------------------------------------------------------------------------
// 2) warp_nearest (frames 0,2) + LayerNorm -> f16 NHWC kvn[(img*HW+p)*64 + c]
// ---------------------------------------------------------------------------
__global__ void prep_kv_kernel(const float* __restrict__ k_inp,
                               const float* __restrict__ flow_f,
                               const float* __restrict__ flow_b,
                               const float* __restrict__ g,
                               const float* __restrict__ b,
                               _Float16* __restrict__ kvn) {
  int idx = blockIdx.x * blockDim.x + threadIdx.x;   // (n, f, p): 2*3*16384
  int p = idx & (HW - 1);
  int nf = idx >> 14;
  int f = nf % 3;
  int n = nf / 3;
  int y = p >> 7, x = p & 127;

  int sx = x, sy = y;
  float validf = 1.0f;
  if (f != 1) {
    const float* fl = (f == 0) ? flow_f : flow_b;
    float fx = fl[((size_t)n * 2 + 0) * HW + p];
    float fy = fl[((size_t)n * 2 + 1) * HW + p];
    int ix = (int)rintf((float)x + fx);   // round-half-even == jnp.round
    int iy = (int)rintf((float)y + fy);
    bool valid = (ix >= 0) && (ix < WW) && (iy >= 0) && (iy < HH);
    validf = valid ? 1.0f : 0.0f;
    sx = min(max(ix, 0), WW - 1);
    sy = min(max(iy, 0), HH - 1);
  }
  const float* base = k_inp + ((size_t)(n * 3 + f)) * CIN * HW + (sy * WW + sx);
  float vals[CIN];
  float mu = 0.f;
#pragma unroll 8
  for (int c = 0; c < CIN; ++c) {
    vals[c] = base[(size_t)c * HW] * validf;
    mu += vals[c];
  }
  mu *= (1.0f / CIN);
  float var = 0.f;
#pragma unroll 8
  for (int c = 0; c < CIN; ++c) { float d = vals[c] - mu; var += d * d; }
  var *= (1.0f / CIN);
  float rs = rsqrtf(var + 1e-5f);
  _Float16* ob = kvn + ((size_t)(n * 3 + f) * HW + p) * CIN;
#pragma unroll 8
  for (int c = 0; c < CIN; ++c)
    ob[c] = (_Float16)((vals[c] - mu) * rs * g[c] + b[c]);
}

// ---------------------------------------------------------------------------
// 3) fused conv_q + conv_kv + attention. One workgroup (8 waves) per window.
// ---------------------------------------------------------------------------
__global__ __launch_bounds__(256)
void fused_attn_kernel(const _Float16* __restrict__ qn,
                       const _Float16* __restrict__ kvn,
                       const _Float16* __restrict__ WqH,   // (512, 9, 64)
                       const _Float16* __restrict__ WkvH,  // (1024, 9, 64)
                       const float* __restrict__ static_a,
                       _Float16* __restrict__ Obuf) {
  // LDS: q 16x512, k 48x512, v 48x512 (f16) = 112KB of the 320KB WGP LDS
  __shared__ _Float16 qs[QL * INNER];
  __shared__ _Float16 ks[3 * QL * INNER];
  __shared__ _Float16 vs[3 * QL * INNER];

  int win  = blockIdx.x;               // 0..2047
  int n    = win >> 10;
  int wloc = win & (NWIN - 1);
  int wy = wloc >> 5, wx = wloc & 31;

  int wv   = threadIdx.x >> 5;         // wave 0..7
  int lane = threadIdx.x & 31;
  int m    = lane & 15;
  int nn   = lane & 15;
  int half = lane >> 4;
  int hoff = half << 3;
  int boff = half << 4;
  int y = wy * 4 + (m >> 2), x = wx * 4 + (m & 3);

  // ---------------- conv phase: 56 tiles of (M=16 pixels, N=64 ch) ---------
  for (int i = 0; i < 7; ++i) {
    int tile = wv * 7 + i;             // 0..55, uniform per wave
    const _Float16* img;
    const _Float16* Wrow;
    _Float16* dst;
    int wco0, dco0;
    float scale;
    if (tile < 8) {                    // q: 8 tiles x 64 = 512 channels
      img = qn + (size_t)n * HW * CIN;
      Wrow = WqH;
      wco0 = tile * 64; dco0 = wco0;
      dst = qs; scale = 0.125f;        // fold DIM_HEAD^-0.5 into q
    } else {                           // kv: 3 frames x 16 tiles x 64 = 1024
      int idx = tile - 8;
      int f = idx >> 4, nt = idx & 15;
      img = kvn + (size_t)(n * 3 + f) * HW * CIN;
      Wrow = WkvH;
      wco0 = nt * 64;
      dco0 = wco0 & (INNER - 1);
      dst = ((wco0 < INNER) ? ks : vs) + f * QL * INNER;
      scale = 1.0f;
    }

    v8f acc[4];
#pragma unroll
    for (int t = 0; t < 4; ++t)
#pragma unroll
      for (int r = 0; r < 8; ++r) acc[t][r] = 0.f;

#pragma unroll
    for (int tap = 0; tap < 9; ++tap) {
      const int dy = tap / 3 - 1, dx = tap % 3 - 1;
      int yy = y + dy, xx = x + dx;
      bool inb = (yy >= 0) && (yy < HH) && (xx >= 0) && (xx < WW);
      const _Float16* px = img + (size_t)(yy * WW + xx) * CIN;
#pragma unroll
      for (int hf = 0; hf < 2; ++hf) {
        v16h a = {};
        if (inb) a = load_a_runs(px + hf * 32 + boff);
        const _Float16* wb = Wrow + tap * CIN + hf * 32 + boff;
#pragma unroll
        for (int t = 0; t < 4; ++t) {
          v16h b = *(const v16h*)(wb + (size_t)(wco0 + t * 16 + nn) * (CIN * 9));
          acc[t] = wmma_f16(a, b, acc[t]);
        }
      }
    }
#pragma unroll
    for (int t = 0; t < 4; ++t)
#pragma unroll
      for (int r = 0; r < 8; ++r) {
        int mm = r + half * 8;
        dst[mm * INNER + dco0 + t * 16 + nn] = (_Float16)(acc[t][r] * scale);
      }
  }
  __syncthreads();

  // ---------------- attention phase: wave = head ---------------------------
  int hd = wv;

  // q fragments from LDS (aligned 16B ds reads)
  const _Float16* qp = qs + (size_t)(lane & 15) * INNER + hd * DHEAD;
  v16h qa[2];
#pragma unroll
  for (int h2 = 0; h2 < 2; ++h2) qa[h2] = load_a_runs(qp + h2 * 32 + boff);

  // sim tiles: 3 frames x (16q x 16k)
  v8f sim[3];
#pragma unroll
  for (int t = 0; t < 3; ++t) {
    const _Float16* kp = ks + ((size_t)(t * QL + nn)) * INNER
                       + hd * DHEAD + (half << 4);
#pragma unroll
    for (int r = 0; r < 8; ++r) sim[t][r] = 0.f;
#pragma unroll
    for (int h2 = 0; h2 < 2; ++h2) {
      v16h bk = *(const v16h*)(kp + h2 * 32);
      sim[t] = wmma_f16(qa[h2], bk, sim[t]);
    }
#pragma unroll
    for (int r = 0; r < 8; ++r) {
      int mm = r + half * 8;
      sim[t][r] += static_a[(hd * QL + mm) * KVL + t * 16 + nn];
    }
  }

  // q region is dead for all waves after this point; reuse as attn staging
  __syncthreads();
  _Float16* atile = qs + wv * (QL * 64);

  // softmax per query row (row m lives in one 16-lane half)
#pragma unroll
  for (int r = 0; r < 8; ++r) {
    float mx = fmaxf(sim[0][r], fmaxf(sim[1][r], sim[2][r]));
    mx = fmaxf(mx, __shfl_xor(mx, 8));
    mx = fmaxf(mx, __shfl_xor(mx, 4));
    mx = fmaxf(mx, __shfl_xor(mx, 2));
    mx = fmaxf(mx, __shfl_xor(mx, 1));
    float e0 = __expf(sim[0][r] - mx);
    float e1 = __expf(sim[1][r] - mx);
    float e2 = __expf(sim[2][r] - mx);
    float s = e0 + e1 + e2;
    s += __shfl_xor(s, 8);
    s += __shfl_xor(s, 4);
    s += __shfl_xor(s, 2);
    s += __shfl_xor(s, 1);
    float inv = __frcp_rn(s);
    int mm = r + half * 8;
    atile[mm * 64 +  0 + nn] = (_Float16)(e0 * inv);
    atile[mm * 64 + 16 + nn] = (_Float16)(e1 * inv);
    atile[mm * 64 + 32 + nn] = (_Float16)(e2 * inv);
    atile[mm * 64 + 48 + nn] = (_Float16)0.f;   // K pad 48..63
  }

  // reload attn as A fragments (own wave's LDS slice; no barrier needed)
  v16h aa[2];
#pragma unroll
  for (int h2 = 0; h2 < 2; ++h2)
    aa[h2] = load_a_runs(atile + (lane & 15) * 64 + h2 * 32 + boff);

  // out = attn @ v : 4 N-tiles of 16 over d
#pragma unroll
  for (int t = 0; t < 4; ++t) {
    v8f oc;
#pragma unroll
    for (int r = 0; r < 8; ++r) oc[r] = 0.f;
#pragma unroll
    for (int h2 = 0; h2 < 2; ++h2) {
      int jbase = h2 * 32 + (half << 4);       // 16 consecutive j per run
      v16h bv = {};
      if (jbase < KVL) {
        const _Float16* vp = vs + (size_t)jbase * INNER
                           + hd * DHEAD + t * 16 + nn;
#pragma unroll
        for (int e = 0; e < 16; ++e) bv[e] = vp[(size_t)e * INNER];
      }
      oc = wmma_f16(aa[h2], bv, oc);
    }
    size_t ob = (size_t)win * QL * INNER;
#pragma unroll
    for (int r = 0; r < 8; ++r) {
      int mm = r + half * 8;
      Obuf[ob + (size_t)mm * INNER + hd * DHEAD + t * 16 + nn] = (_Float16)oc[r];
    }
  }
}

// ---------------------------------------------------------------------------
// 4) conv_out: Obuf(windowed, channel-contiguous) * Wout(64 x 9*512) -> fp32
// one wave per window (N=64); 2048 waves; K = 9 taps x 16 chunks of 32
// ---------------------------------------------------------------------------
__global__ void conv_out_kernel(const _Float16* __restrict__ Obuf,
                                const _Float16* __restrict__ Wh,
                                float* __restrict__ out) {
  int wave = blockIdx.x * (blockDim.x >> 5) + (threadIdx.x >> 5);
  int lane = threadIdx.x & 31;
  int wloc = wave & (NWIN - 1);
  int n    = wave >> 10;
  int wy = wloc >> 5, wx = wloc & 31;
  int m = lane & 15;
  int hoff = (lane >> 4) << 3;
  int boff = (lane >> 4) << 4;
  int nn = lane & 15;
  int y = wy * 4 + (m >> 2), x = wx * 4 + (m & 3);

  v8f acc[4];
#pragma unroll
  for (int t = 0; t < 4; ++t)
#pragma unroll
    for (int r = 0; r < 8; ++r) acc[t][r] = 0.f;

#pragma unroll
  for (int tap = 0; tap < 9; ++tap) {
    const int dy = tap / 3 - 1, dx = tap % 3 - 1;
    int yy = y + dy, xx = x + dx;
    bool inb = (yy >= 0) && (yy < HH) && (xx >= 0) && (xx < WW);
    int wl = ((yy >> 2) << 5) + (xx >> 2);
    int ii = ((yy & 3) << 2) + (xx & 3);
    const _Float16* px = Obuf + ((size_t)(n * NWIN + wl) * QL + ii) * INNER;
    const _Float16* wt = Wh + tap * INNER;
    for (int c0 = 0; c0 < INNER; c0 += 32) {
      v16h a = {};
      if (inb) a = load_a_runs(px + c0 + boff);
      const _Float16* wb = wt + c0 + boff;
#pragma unroll
      for (int t = 0; t < 4; ++t) {
        v16h b = *(const v16h*)(wb + (size_t)(t * 16 + nn) * (INNER * 9));
        acc[t] = wmma_f16(a, b, acc[t]);
      }
    }
  }
  int half = lane >> 4;
#pragma unroll
  for (int t = 0; t < 4; ++t)
#pragma unroll
    for (int r = 0; r < 8; ++r) {
      int mm = r + half * 8;
      int yy = wy * 4 + (mm >> 2), xx = wx * 4 + (mm & 3);
      int co = t * 16 + nn;
      out[((size_t)n * CIN + co) * HW + yy * WW + xx] = acc[t][r];
    }
}

// ---------------------------------------------------------------------------
// launch
// ---------------------------------------------------------------------------
extern "C" void kernel_launch(void* const* d_in, const int* in_sizes, int n_in,
                              void* d_out, int out_size, void* d_ws, size_t ws_size,
                              hipStream_t stream) {
  (void)in_sizes; (void)n_in; (void)out_size; (void)ws_size;
  const float* q_inp    = (const float*)d_in[0];
  const float* k_inp    = (const float*)d_in[1];
  const float* flow_f   = (const float*)d_in[2];
  const float* flow_b   = (const float*)d_in[3];
  const float* gq       = (const float*)d_in[4];
  const float* bq       = (const float*)d_in[5];
  const float* gkv      = (const float*)d_in[6];
  const float* bkv      = (const float*)d_in[7];
  const float* Wq       = (const float*)d_in[8];
  const float* Wkv      = (const float*)d_in[9];
  const float* Wout     = (const float*)d_in[10];
  const float* static_a = (const float*)d_in[11];
  float* out = (float*)d_out;

  // ws carve-up (f16 elements); total ~= 53 MB
  _Float16* ws = (_Float16*)d_ws;
  size_t off = 0;
  _Float16* qn    = ws + off; off += (size_t)2 * HW * CIN;
  _Float16* kvn   = ws + off; off += (size_t)6 * HW * CIN;
  _Float16* WqH   = ws + off; off += (size_t)INNER * CIN * 9;
  _Float16* WkvH  = ws + off; off += (size_t)2 * INNER * CIN * 9;
  _Float16* WoH   = ws + off; off += (size_t)CIN * INNER * 9;
  _Float16* Obuf  = ws + off; off += (size_t)2048 * QL * INNER;

  // 0) weights -> f16, tap-major K
  {
    int n0 = INNER * CIN * 9;
    repack_w_kernel<<<(n0 + 255) / 256, 256, 0, stream>>>(Wq, WqH, INNER, CIN);
    int n1 = 2 * INNER * CIN * 9;
    repack_w_kernel<<<(n1 + 255) / 256, 256, 0, stream>>>(Wkv, WkvH, 2 * INNER, CIN);
    int n2 = CIN * INNER * 9;
    repack_w_kernel<<<(n2 + 255) / 256, 256, 0, stream>>>(Wout, WoH, CIN, INNER);
  }
  prep_q_kernel<<<(2 * HW) / 256, 256, 0, stream>>>(q_inp, gq, bq, qn);
  prep_kv_kernel<<<(6 * HW) / 256, 256, 0, stream>>>(k_inp, flow_f, flow_b,
                                                     gkv, bkv, kvn);
  // fused conv_q + conv_kv + attention: one workgroup per window
  fused_attn_kernel<<<2048, 256, 0, stream>>>(qn, kvn, WqH, WkvH, static_a, Obuf);
  // output conv (needs cross-window halo -> separate kernel)
  conv_out_kernel<<<512, 128, 0, stream>>>(Obuf, WoH, out);
}
